// DynamicEdge_76098230550961
// MI455X (gfx1250) — compile-verified, matrix operand order checked by reading, not run on hardware
//
#include <hip/hip_runtime.h>

#define NPTS 8192
#define KNN  32
#define BN_INV_C 0.9999950000374997f

typedef __attribute__((ext_vector_type(16))) __bf16 v16bf;
typedef __attribute__((ext_vector_type(8)))  __bf16 v8bf;
typedef __attribute__((ext_vector_type(8)))  float  v8f;

// ---------------- fragment helpers (16-bit A/B VGPR layout, wave32) ----------
// Lane l: half = l>>4. Vector elements e=0..7 hold K = base + half*8 + e,
// elements e=8..15 hold K = base + 16 + half*8 + (e-8).  (ISA 7.12.2)

__device__ __forceinline__ v8bf load8(const __bf16* p) {
  return *(const v8bf*)p;   // 16-byte aligned by construction
}

__device__ __forceinline__ v16bf combine8(v8bf lo, v8bf hi) {
  v16bf r;
#pragma unroll
  for (int e = 0; e < 8; ++e) { r[e] = lo[e]; r[8 + e] = hi[e]; }
  return r;
}

// Row-major source: row pointer, K-chunk index kc (32 elements), lane half.
__device__ __forceinline__ v16bf load_frag_row(const __bf16* row, int kc, int half) {
  const __bf16* p = row + kc * 32 + half * 8;
  return combine8(load8(p), load8(p + 16));
}

// (x_j - x_i) fragment for EdgeConv second half of the concat feature.
__device__ __forceinline__ v16bf diff_frag(const __bf16* xj, const __bf16* xi,
                                           int kc, int half) {
  int o = kc * 32 + half * 8;
  v8bf jl = load8(xj + o),      il = load8(xi + o);
  v8bf jh = load8(xj + o + 16), ih = load8(xi + o + 16);
  v16bf r;
#pragma unroll
  for (int e = 0; e < 8; ++e) {
    r[e]     = (__bf16)((float)jl[e] - (float)il[e]);
    r[8 + e] = (__bf16)((float)jh[e] - (float)ih[e]);
  }
  return r;
}

__device__ __forceinline__ v8f wmma_bf16(v16bf a, v16bf b, v8f c) {
  return __builtin_amdgcn_wmma_f32_16x16x32_bf16(
      false, a, false, b, (short)0, c, false, false);
}

// ---------------- prep kernels ----------------------------------------------

__global__ void cvt_bf16_kernel(const float* __restrict__ in,
                                __bf16* __restrict__ out, int n) {
  int t = blockIdx.x * blockDim.x + threadIdx.x;
  if (t < n) out[t] = (__bf16)in[t];
}

// WT[n*K + k] = W[k*N + n], zero-padded to Npad columns (bf16)
__global__ void wtrans_kernel(const float* __restrict__ W,
                              __bf16* __restrict__ WT, int K, int N, int Npad) {
  int t = blockIdx.x * blockDim.x + threadIdx.x;
  if (t < Npad * K) {
    int n = t / K, k = t % K;
    WT[t] = (__bf16)((n < N) ? W[(size_t)k * N + n] : 0.0f);
  }
}

__global__ void rownorm2_kernel(const __bf16* __restrict__ X,
                                float* __restrict__ d2, int C) {
  int r = blockIdx.x * blockDim.x + threadIdx.x;
  if (r < NPTS) {
    const __bf16* row = X + (size_t)r * C;
    float s = 0.f;
    for (int c = 0; c < C; ++c) { float v = (float)row[c]; s += v * v; }
    d2[r] = s;
  }
}

// ---------------- fused kNN: WMMA distance tiles + online top-32 ------------
// Block: 512 threads = 16 waves, handles 16 query rows. Wave w produces one
// 16x16 dot tile per 256-column chunk (phase A), wave r maintains row r's
// top-32 as one (val,idx) per lane (phase B).

template <int C>
__global__ __launch_bounds__(512) void knn_kernel(const __bf16* __restrict__ xh,
                                                  const float* __restrict__ d2,
                                                  int* __restrict__ idx_out) {
  constexpr int KCH = C / 32;
  const int lane = threadIdx.x & 31;
  const int wave = threadIdx.x >> 5;   // 0..15
  const int half = lane >> 4;
  const int ln16 = lane & 15;
  const int rowbase = blockIdx.x * 16;

  __shared__ float sc[16 * 257];       // 16 rows x 256 cols, stride 257

  // A fragments: rows rowbase+0..15, resident for the whole kernel
  v16bf afr[KCH];
  {
    const __bf16* arow = xh + (size_t)(rowbase + ln16) * C;
#pragma unroll
    for (int kc = 0; kc < KCH; ++kc) afr[kc] = load_frag_row(arow, kc, half);
  }

  float lval = 3.0e38f;   // this lane's slot of the row's top-32 list
  int   lidx = 0;
  float kth  = 3.0e38f;   // current 32nd-smallest (max of list), wave-uniform

  for (int cb = 0; cb < NPTS / 256; ++cb) {
    // -------- phase A: 16 waves each compute a 16x16 dot tile -------------
    const int j0 = cb * 256 + wave * 16;
    const __bf16* brow = xh + (size_t)(j0 + ln16) * C;
    if (j0 + 256 < NPTS)  // warm next chunk's B rows into cache
      __builtin_prefetch(xh + (size_t)(j0 + 256 + ln16) * C, 0, 1);
    v8f acc = {};
#pragma unroll
    for (int kc = 0; kc < KCH; ++kc)
      acc = wmma_bf16(afr[kc], load_frag_row(brow, kc, half), acc);

    const float d2j = d2[j0 + ln16];
#pragma unroll
    for (int g = 0; g < 8; ++g) {
      int M = g + 8 * half;                       // D layout: M = g + 8*half
      float s = d2j - 2.0f * acc[g];              // d2[i] const per row: drop
      if (j0 + ln16 == rowbase + M) s = 3.2e38f;  // exclude self-loop
      sc[M * 257 + wave * 16 + ln16] = s;
    }
    __syncthreads();

    // -------- phase B: wave r scans row r's 256 candidates ----------------
#pragma unroll
    for (int t = 0; t < 8; ++t) {
      int col = lane + 32 * t;
      float v = sc[wave * 257 + col];
      int   j = cb * 256 + col;
      bool want = v < kth;
      unsigned long long m;
      while ((m = __ballot(want)) != 0ull) {
        int src = __ffsll(m) - 1;
        float cv = __shfl(v, src);
        int   cj = __shfl(j, src);
        // find lane holding the current list maximum
        float mv = lval; int ml = lane;
#pragma unroll
        for (int off = 16; off; off >>= 1) {
          float ov = __shfl_xor(mv, off);
          int   ol = __shfl_xor(ml, off);
          if (ov > mv || (ov == mv && ol > ml)) { mv = ov; ml = ol; }
        }
        if (lane == ml) { lval = cv; lidx = cj; }   // evict max, insert cand.
        float nk = lval;                             // refresh kth
#pragma unroll
        for (int off = 16; off; off >>= 1) nk = fmaxf(nk, __shfl_xor(nk, off));
        kth = nk;
        if (lane == src) want = false;
        want = want && (v < kth);
      }
    }
    __syncthreads();
  }
  // max-aggregation downstream is order-invariant: unsorted indices are fine
  idx_out[(size_t)(rowbase + wave) * KNN + lane] = lidx;
}

// ---------------- EdgeConv: gather + concat-diff GEMM + ReLU/BN + max -------
// Block = 1 point, 256 threads = 8 waves (one 16-wide N-tile each), two
// M-tiles covering the 32 neighbors. Output channels = 128 for both convs.

template <int CIN>
__global__ __launch_bounds__(256) void edgeconv_kernel(
    const __bf16* __restrict__ xin, const int* __restrict__ idx,
    const __bf16* __restrict__ WT, const float* __restrict__ bias,
    const float* __restrict__ gain, const float* __restrict__ beta,
    __bf16* __restrict__ out) {
  constexpr int KDIM = 2 * CIN;
  constexpr int KCH  = KDIM / 32;
  const int lane = threadIdx.x & 31;
  const int wave = threadIdx.x >> 5;   // N-tile id
  const int half = lane >> 4;
  const int ln16 = lane & 15;
  const int p = blockIdx.x;

  const int j0 = idx[p * KNN + ln16];        // neighbor rows, M-tile 0
  const int j1 = idx[p * KNN + 16 + ln16];   // M-tile 1
  const __bf16* xi  = xin + (size_t)p  * CIN;
  const __bf16* xj0 = xin + (size_t)j0 * CIN;
  const __bf16* xj1 = xin + (size_t)j1 * CIN;
  const __bf16* wrow = WT + (size_t)(wave * 16 + ln16) * KDIM;

  v8f acc0 = {}, acc1 = {};
#pragma unroll
  for (int kc = 0; kc < KCH; ++kc) {
    v16bf bf = load_frag_row(wrow, kc, half);
    if (kc < CIN / 32) {                       // first half of concat: x_i
      v16bf ax = load_frag_row(xi, kc, half);  // identical for all neighbors
      acc0 = wmma_bf16(ax, bf, acc0);
      acc1 = wmma_bf16(ax, bf, acc1);
    } else {                                   // second half: x_j - x_i
      int kc2 = kc - CIN / 32;
      acc0 = wmma_bf16(diff_frag(xj0, xi, kc2, half), bf, acc0);
      acc1 = wmma_bf16(diff_frag(xj1, xi, kc2, half), bf, acc1);
    }
  }

  const int n = wave * 16 + ln16;
  const float b  = bias[n];
  const float s  = gain[n] * BN_INV_C;
  const float be = beta[n];
  float pm = -3.0e38f;
#pragma unroll
  for (int g = 0; g < 8; ++g) {
    float z0 = fmaxf(acc0[g] + b, 0.0f) * s + be;
    float z1 = fmaxf(acc1[g] + b, 0.0f) * s + be;
    pm = fmaxf(pm, fmaxf(z0, z1));
  }
  pm = fmaxf(pm, __shfl_xor(pm, 16));          // merge the two row halves
  if (half == 0) out[(size_t)p * 128 + n] = (__bf16)pm;
}

// ---------------- generic WMMA GEMM for the MLP tail ------------------------
// Y = act(X @ W + b); X: MxK bf16 row-major, WT: Npad x K bf16 (col-contig).

__global__ __launch_bounds__(256) void gemm_kernel(
    const __bf16* __restrict__ X, const __bf16* __restrict__ WT,
    const float* __restrict__ bias, const float* __restrict__ gain,
    const float* __restrict__ beta, float* __restrict__ outf,
    __bf16* __restrict__ outb, int M, int K, int Npad, int Nreal, int relu_bn) {
  const int lane = threadIdx.x & 31;
  const int wave = threadIdx.x >> 5;
  const int half = lane >> 4, ln16 = lane & 15;
  const int nTiles = Npad >> 4;
  const int tile = blockIdx.x * 8 + wave;
  if (tile >= (M >> 4) * nTiles) return;       // wave-uniform: EXEC stays full
  const int mT = tile / nTiles, nT = tile % nTiles;

  const __bf16* xrow = X  + (size_t)(mT * 16 + ln16) * K;
  const __bf16* wrow = WT + (size_t)(nT * 16 + ln16) * K;
  v8f acc = {};
  for (int kc = 0; kc < (K >> 5); ++kc)
    acc = wmma_bf16(load_frag_row(xrow, kc, half),
                    load_frag_row(wrow, kc, half), acc);

  const int n = nT * 16 + ln16;
  if (n < Nreal) {
    float b  = bias[n];
    float s  = relu_bn ? gain[n] * BN_INV_C : 1.0f;
    float be = relu_bn ? beta[n] : 0.0f;
#pragma unroll
    for (int g = 0; g < 8; ++g) {
      int m = mT * 16 + g + 8 * half;
      float z = acc[g] + b;
      float v = relu_bn ? (fmaxf(z, 0.0f) * s + be) : z;
      if (outf) outf[(size_t)m * Nreal + n] = v;
      else      outb[(size_t)m * Nreal + n] = (__bf16)v;
    }
  }
}

// ---------------- host side -------------------------------------------------

extern "C" void kernel_launch(void* const* d_in, const int* in_sizes, int n_in,
                              void* d_out, int out_size, void* d_ws, size_t ws_size,
                              hipStream_t stream) {
  (void)in_sizes; (void)n_in; (void)out_size; (void)ws_size;
  const float* x   = (const float*)d_in[0];
  const float* W1  = (const float*)d_in[1];  const float* b1  = (const float*)d_in[2];
  const float* g1  = (const float*)d_in[3];  const float* be1 = (const float*)d_in[4];
  const float* W2  = (const float*)d_in[5];  const float* b2  = (const float*)d_in[6];
  const float* g2  = (const float*)d_in[7];  const float* be2 = (const float*)d_in[8];
  const float* W3  = (const float*)d_in[9];  const float* b3  = (const float*)d_in[10];
  const float* g3  = (const float*)d_in[11]; const float* be3 = (const float*)d_in[12];
  const float* W4  = (const float*)d_in[13]; const float* b4  = (const float*)d_in[14];
  const float* g4  = (const float*)d_in[15]; const float* be4 = (const float*)d_in[16];
  const float* W5  = (const float*)d_in[17]; const float* b5  = (const float*)d_in[18];
  const float* g5  = (const float*)d_in[19]; const float* be5 = (const float*)d_in[20];
  const float* W6  = (const float*)d_in[21]; const float* b6  = (const float*)d_in[22];

  char* wp = (char*)d_ws;
  auto alloc = [&](size_t bytes) -> char* {
    char* p = wp; wp += (bytes + 255) & ~(size_t)255; return p;
  };
  __bf16* xh  = (__bf16*)alloc((size_t)NPTS * 64  * 2);
  float*  d2x = (float*) alloc((size_t)NPTS * 4);
  int*    idx1= (int*)   alloc((size_t)NPTS * KNN * 4);
  __bf16* h1  = (__bf16*)alloc((size_t)NPTS * 128 * 2);
  float*  d2h = (float*) alloc((size_t)NPTS * 4);
  int*    idx2= (int*)   alloc((size_t)NPTS * KNN * 4);
  __bf16* h2  = (__bf16*)alloc((size_t)NPTS * 128 * 2);
  __bf16* h3  = (__bf16*)alloc((size_t)NPTS * 256 * 2);
  __bf16* h4  = (__bf16*)alloc((size_t)NPTS * 128 * 2);
  __bf16* h5  = (__bf16*)alloc((size_t)NPTS * 64  * 2);
  __bf16* W1T = (__bf16*)alloc(128 * 128 * 2);
  __bf16* W2T = (__bf16*)alloc(128 * 256 * 2);
  __bf16* W3T = (__bf16*)alloc(256 * 128 * 2);
  __bf16* W4T = (__bf16*)alloc(128 * 256 * 2);
  __bf16* W5T = (__bf16*)alloc(64  * 128 * 2);
  __bf16* W6T = (__bf16*)alloc(48  * 64  * 2);   // N padded 40 -> 48

  // prep: bf16 conversion, norms, weight transposes
  cvt_bf16_kernel<<<(NPTS * 64 + 255) / 256, 256, 0, stream>>>(x, xh, NPTS * 64);
  rownorm2_kernel<<<(NPTS + 255) / 256, 256, 0, stream>>>(xh, d2x, 64);
  wtrans_kernel<<<(128 * 128 + 255) / 256, 256, 0, stream>>>(W1, W1T, 128, 128, 128);
  wtrans_kernel<<<(128 * 256 + 255) / 256, 256, 0, stream>>>(W2, W2T, 256, 128, 128);
  wtrans_kernel<<<(256 * 128 + 255) / 256, 256, 0, stream>>>(W3, W3T, 128, 256, 256);
  wtrans_kernel<<<(128 * 256 + 255) / 256, 256, 0, stream>>>(W4, W4T, 256, 128, 128);
  wtrans_kernel<<<(64 * 128 + 255) / 256, 256, 0, stream>>>(W5, W5T, 128, 64, 64);
  wtrans_kernel<<<(48 * 64 + 255) / 256, 256, 0, stream>>>(W6, W6T, 64, 40, 48);

  // dynamic graph 1 + EdgeConv1
  knn_kernel<64><<<NPTS / 16, 512, 0, stream>>>(xh, d2x, idx1);
  edgeconv_kernel<64><<<NPTS, 256, 0, stream>>>(xh, idx1, W1T, b1, g1, be1, h1);

  // dynamic graph 2 + EdgeConv2
  rownorm2_kernel<<<(NPTS + 255) / 256, 256, 0, stream>>>(h1, d2h, 128);
  knn_kernel<128><<<NPTS / 16, 512, 0, stream>>>(h1, d2h, idx2);
  edgeconv_kernel<128><<<NPTS, 256, 0, stream>>>(h1, idx2, W2T, b2, g2, be2, h2);

  // MLP tail: tiles = (M/16)*(Npad/16), 8 wave-tiles per block
  gemm_kernel<<<(512 * 16) / 8, 256, 0, stream>>>(h2, W3T, b3, g3, be3,
      nullptr, h3, NPTS, 128, 256, 256, 1);
  gemm_kernel<<<(512 * 8) / 8, 256, 0, stream>>>(h3, W4T, b4, g4, be4,
      nullptr, h4, NPTS, 256, 128, 128, 1);
  gemm_kernel<<<(512 * 4) / 8, 256, 0, stream>>>(h4, W5T, b5, g5, be5,
      nullptr, h5, NPTS, 128, 64, 64, 1);
  gemm_kernel<<<(512 * 3) / 8, 256, 0, stream>>>(h5, W6T, b6, nullptr, nullptr,
      (float*)d_out, nullptr, NPTS, 64, 48, 40, 0);
}
